// DGM_d_26328149524712
// MI455X (gfx1250) — matrix operand order repaired and checked by reading, not compiled, not run on hardware
//
#include <hip/hip_runtime.h>
#include <hip/hip_bf16.h>
#include <math.h>

// Problem constants (from reference)
#define NN    8192   // nodes
#define DIN   512
#define DEMB  256
#define KNBR  16     // neighbors per node

typedef __attribute__((ext_vector_type(16))) __bf16 v16bf;
typedef __attribute__((ext_vector_type(8)))  __bf16 v8bf;
typedef __attribute__((ext_vector_type(8)))  float  v8f;
typedef __attribute__((ext_vector_type(2)))  float  v2f;

// ---------------------------------------------------------------------------
// Small transpose: Wt[n][k] = W[k][n]   (512x256 -> 256x512)
// ---------------------------------------------------------------------------
__global__ void __launch_bounds__(256) transposeW_kernel(const float* __restrict__ W,
                                                         float* __restrict__ Wt) {
    int idx = blockIdx.x * 256 + threadIdx.x;    // 0 .. 512*256-1
    int k = idx >> 8;          // 0..511
    int n = idx & 255;         // 0..255
    Wt[(size_t)n * DIN + k] = W[idx];
}

// ---------------------------------------------------------------------------
// Generic bf16-WMMA GEMM:  C[M][256] = (relu?) Amat[M][Kdim] @ Bt[256][Kdim]^T
// One WG (8 waves) per 16-row strip; each wave owns 2 n-tiles.
// A tile staged through LDS (f32 -> bf16, non-temporal loads: A is streamed
// once with no cross-WG reuse); B fragments gathered from global
// (contiguous 16-float runs, L2-resident operand).
// Compile-time RELU/STORET/DUAL -> branch-free epilogue.
// ---------------------------------------------------------------------------
template <bool RELU, bool STORET, bool DUAL>
__global__ void __launch_bounds__(256) gemm_bf16_wmma(
    const float* __restrict__ Amat,
    const float* __restrict__ Bt,
    float* __restrict__ C,
    float* __restrict__ C2,
    int M, int Kdim)
{
    __shared__ __bf16 As[16][40];   // pitch 40 bf16 = 80B (16B aligned rows)

    const int tid  = threadIdx.x;
    const int wave = tid >> 5;
    const int lane = tid & 31;
    const int half = lane >> 4;
    const int lrow = lane & 15;
    const int m0   = blockIdx.x * 16;

    v8f z = {0.f,0.f,0.f,0.f,0.f,0.f,0.f,0.f};
    v8f acc0 = z, acc1 = z;

    for (int k0 = 0; k0 < Kdim; k0 += 32) {
        // --- cooperative A staging: 16 rows x 32 cols, f32 -> bf16 (NT load) ---
        {
            int r = tid >> 4;            // 0..15
            int c = (tid & 15) * 2;      // 0..30 step 2
            const v2f* ap = (const v2f*)(Amat + (size_t)(m0 + r) * Kdim + k0 + c);
            v2f f = __builtin_nontemporal_load(ap);
            As[r][c]     = (__bf16)f.x;
            As[r][c + 1] = (__bf16)f.y;
        }
        __syncthreads();

        // --- A fragment: lane lrow holds row m0+lrow; two runs of 8 K ---
        const __bf16* arow = &As[lrow][0];
        v8bf alo = *(const v8bf*)(arow + half * 16);
        v8bf ahi = *(const v8bf*)(arow + half * 16 + 8);
        v16bf afrag = __builtin_shufflevector(alo, ahi,
            0,1,2,3,4,5,6,7,8,9,10,11,12,13,14,15);

        // --- B fragments from global: n-tiles (wave) and (wave+8) ---
        #pragma unroll
        for (int t = 0; t < 2; ++t) {
            int nt = wave + t * 8;
            int n  = nt * 16 + lrow;
            const float* bp = Bt + (size_t)n * Kdim + k0 + half * 16;
            v16bf bfrag;
            #pragma unroll
            for (int i = 0; i < 16; ++i) bfrag[i] = (__bf16)bp[i];
            if (t == 0)
                acc0 = __builtin_amdgcn_wmma_f32_16x16x32_bf16(
                    false, afrag, false, bfrag, (short)0, acc0, false, false);
            else
                acc1 = __builtin_amdgcn_wmma_f32_16x16x32_bf16(
                    false, afrag, false, bfrag, (short)0, acc1, false, false);
        }
        __syncthreads();
    }

    // --- branch-free epilogue ---
    #pragma unroll
    for (int t = 0; t < 2; ++t) {
        int nt = wave + t * 8;
        int n  = nt * 16 + lrow;
        #pragma unroll
        for (int v = 0; v < 8; ++v) {
            float val = (t == 0) ? acc0[v] : acc1[v];
            if (RELU) val = fmaxf(val, 0.0f);
            int m = m0 + v + half * 8;
            if (STORET) {
                C[(size_t)n * M + m] = val;
            } else {
                size_t o = (size_t)m * DEMB + n;
                C[o] = val;
                if (DUAL) C2[o] = val;
            }
        }
    }
}

// ---------------------------------------------------------------------------
// Row squared-norms: sq[i] = sum_d h[i][d]^2.  One wave per row.
// ---------------------------------------------------------------------------
__global__ void __launch_bounds__(256) rownorm_kernel(const float* __restrict__ h,
                                                      float* __restrict__ sq) {
    int wave = threadIdx.x >> 5;
    int lane = threadIdx.x & 31;
    int row  = blockIdx.x * 8 + wave;
    const float* p = h + (size_t)row * DEMB + lane * 8;
    float s = 0.f;
    #pragma unroll
    for (int i = 0; i < 8; ++i) { float v = p[i]; s += v * v; }
    #pragma unroll
    for (int off = 16; off > 0; off >>= 1) s += __shfl_down(s, off, 32);
    if (lane == 0) sq[row] = s;
}

// ---------------------------------------------------------------------------
// Sorted-insert into a 16-entry ascending register-resident list.
// Fully unrolled -> stays in VGPRs (no dynamic indexing).
// ---------------------------------------------------------------------------
__device__ __forceinline__ void insert16(float v, int j, float* bv, int* bi) {
    if (v < bv[15]) {
        bv[15] = v; bi[15] = j;
        #pragma unroll
        for (int s = 15; s > 0; --s) {
            if (bv[s] < bv[s - 1]) {
                float tv = bv[s]; bv[s] = bv[s - 1]; bv[s - 1] = tv;
                int   ti = bi[s]; bi[s] = bi[s - 1]; bi[s - 1] = ti;
            }
        }
    }
}

// ---------------------------------------------------------------------------
// Fused Gram (WMMA) + squared distance + Gumbel perturbation + top-16.
// One WG per 16-row strip. Gram B-operand is h itself (row-major == h^T as Bt).
// Per 128-column chunk: 8 waves each compute one 16x16 lq tile -> LDS,
// then 256 threads (16 lanes per row) fold candidates into register top-16s.
// q is streamed once -> non-temporal loads keep it out of L2.
// ---------------------------------------------------------------------------
__global__ void __launch_bounds__(256) dist_topk_kernel(
    const float* __restrict__ h,     // [NN][DEMB]
    const float* __restrict__ sq,    // [NN]
    const float* __restrict__ q,     // [NN][NN]
    const float* __restrict__ temp,  // scalar
    float* __restrict__ out_edges,   // [2][NN*KNBR] (indices row, then row-ids)
    float* __restrict__ out_logp)    // [NN*KNBR]
{
    __shared__ __bf16 Hs[16][DEMB + 8];     // strip rows, bf16, 528B pitch
    __shared__ float  lqbuf[16][128];
    __shared__ float  mV[16][256];
    __shared__ int    mI[16][256];

    const int tid  = threadIdx.x;
    const int wave = tid >> 5;
    const int lane = tid & 31;
    const int half = lane >> 4;
    const int lrow = lane & 15;
    const int i0   = blockIdx.x * 16;

    const float tc   = fminf(fmaxf(temp[0], -5.0f), 5.0f);
    const float expT = expf(tc);

    // --- stage strip rows h[i0..i0+16) as bf16 ---
    {
        int r = tid >> 4;
        int c = (tid & 15) * 16;
        const float* p = h + (size_t)(i0 + r) * DEMB + c;
        #pragma unroll
        for (int i = 0; i < 16; ++i) Hs[r][c + i] = (__bf16)p[i];
    }
    __syncthreads();

    // --- preload all 8 A fragments (K=256) into registers ---
    v16bf afr[8];
    #pragma unroll
    for (int ks = 0; ks < 8; ++ks) {
        const __bf16* arow = &Hs[lrow][ks * 32];
        v8bf lo = *(const v8bf*)(arow + half * 16);
        v8bf hi = *(const v8bf*)(arow + half * 16 + 8);
        afr[ks] = __builtin_shufflevector(lo, hi,
            0,1,2,3,4,5,6,7,8,9,10,11,12,13,14,15);
    }

    // per-lane row norms for the 8 output rows this lane holds
    float sqi[8];
    #pragma unroll
    for (int v = 0; v < 8; ++v) sqi[v] = sq[i0 + v + half * 8];

    // register top-16 (ascending lq)
    float bv[16]; int bi[16];
    #pragma unroll
    for (int e = 0; e < 16; ++e) { bv[e] = 3.402823466e38f; bi[e] = 0; }

    v8f z = {0.f,0.f,0.f,0.f,0.f,0.f,0.f,0.f};

    for (int ch = 0; ch < NN / 128; ++ch) {
        // --- this wave's 16x16 Gram tile at columns ch*128 + wave*16 ---
        const int jn = ch * 128 + wave * 16 + lrow;
        v8f acc = z;
        #pragma unroll
        for (int ks = 0; ks < 8; ++ks) {
            const float* bp = h + (size_t)jn * DEMB + ks * 32 + half * 16;
            v16bf bfrag;
            #pragma unroll
            for (int i = 0; i < 16; ++i) bfrag[i] = (__bf16)bp[i];
            acc = __builtin_amdgcn_wmma_f32_16x16x32_bf16(
                false, afr[ks], false, bfrag, (short)0, acc, false, false);
        }

        // --- distance + Gumbel, write lq tile to LDS ---
        const float sqj = sq[jn];
        #pragma unroll
        for (int v = 0; v < 8; ++v) {
            int m = i0 + v + half * 8;
            float D = sqi[v] + sqj - 2.0f * acc[v];
            D = fmaxf(D, 0.0f);
            float qv  = __builtin_nontemporal_load(q + (size_t)m * NN + jn);
            float gum = __logf(-__logf(qv + 1e-8f));
            lqbuf[v + half * 8][wave * 16 + lrow] = D * expT - gum;
        }
        __syncthreads();

        // --- fold 128 candidates: 16 lanes per row, 8 candidates each ---
        {
            int r = tid >> 4;          // row within strip
            int l = tid & 15;          // lane group within row
            #pragma unroll
            for (int c = 0; c < 8; ++c) {
                int jl = l * 8 + c;
                insert16(lqbuf[r][jl], ch * 128 + jl, bv, bi);
            }
        }
        __syncthreads();
    }

    // --- merge: dump per-thread lists, one thread per row selects final 16 ---
    {
        int r = tid >> 4;
        int l = tid & 15;
        #pragma unroll
        for (int e = 0; e < 16; ++e) {
            mV[r][l * 16 + e] = bv[e];
            mI[r][l * 16 + e] = bi[e];
        }
    }
    __syncthreads();

    if (tid < 16) {
        int r = tid;
        float fv[16]; int fi[16];
        #pragma unroll
        for (int e = 0; e < 16; ++e) { fv[e] = 3.402823466e38f; fi[e] = 0; }
        for (int c = 0; c < 256; ++c) insert16(mV[r][c], mI[r][c], fv, fi);
        int row = i0 + r;
        #pragma unroll
        for (int e = 0; e < KNBR; ++e) {
            out_logp[(size_t)row * KNBR + e]  = -fv[e];          // descending -lq
            out_edges[(size_t)row * KNBR + e] = (float)fi[e];    // neighbor idx
            out_edges[(size_t)NN * KNBR + (size_t)row * KNBR + e] = (float)row;
        }
    }
}

// ---------------------------------------------------------------------------
// Host launcher
// ---------------------------------------------------------------------------
extern "C" void kernel_launch(void* const* d_in, const int* in_sizes, int n_in,
                              void* d_out, int out_size, void* d_ws, size_t ws_size,
                              hipStream_t stream) {
    const float* x  = (const float*)d_in[0];   // [8192][512]
    const float* A  = (const float*)d_in[1];   // [8192][8192]
    const float* W  = (const float*)d_in[2];   // [512][256]
    const float* T  = (const float*)d_in[3];   // scalar
    const float* q  = (const float*)d_in[4];   // [8192][8192]

    char* ws = (char*)d_ws;
    float* Wt  = (float*)(ws);                                            // 256*512
    float* B1t = (float*)(ws + 512u * 1024u);                             // 256*8192 (xW)^T
    float* hbf = (float*)(ws + 512u * 1024u + 8u * 1024u * 1024u);        // 8192*256
    float* sq  = (float*)(ws + 512u * 1024u + 16u * 1024u * 1024u);       // 8192

    float* hb_out    = (float*)d_out;                        // 8192*256
    float* edges_out = hb_out + (size_t)NN * DEMB;           // 2 * 8192*16
    float* logp_out  = edges_out + 2u * NN * KNBR;           // 8192*16

    // 1) Wt = W^T
    transposeW_kernel<<<dim3(512), dim3(256), 0, stream>>>(W, Wt);

    // 2) B1 = x @ W, stored transposed -> B1t [256][8192]
    gemm_bf16_wmma<false, true, false><<<dim3(NN / 16), dim3(256), 0, stream>>>(
        x, Wt, B1t, nullptr, NN, DIN);

    // 3) h = relu(A @ B1) -> ws and d_out (hb)
    gemm_bf16_wmma<true, false, true><<<dim3(NN / 16), dim3(256), 0, stream>>>(
        A, B1t, hbf, hb_out, NN, NN);

    // 4) row squared norms
    rownorm_kernel<<<dim3(NN / 8), dim3(256), 0, stream>>>(hbf, sq);

    // 5) fused Gram + distances + Gumbel + top-16
    dist_topk_kernel<<<dim3(NN / 16), dim3(256), 0, stream>>>(
        hbf, sq, q, T, edges_out, logp_out);
}
